// Generator_61770219651712
// MI455X (gfx1250) — matrix operand order repaired and checked by reading, not compiled, not run on hardware
//
#include <hip/hip_runtime.h>
#include <hip/hip_bf16.h>
#include <math.h>
#include <type_traits>

// ---------------------------------------------------------------------------
// CDNA5 (gfx1250) generator forward: all convs/matmuls via v_wmma_f32_16x16x32_f16
// Activations kept fp32 in a channel-major "CN" layout: act[c][b*HW + p].
// Weights pre-packed f16, zero-padded to (Mp = ceil64(M)) x (Kp = ceil64(K)).
// ---------------------------------------------------------------------------

typedef __attribute__((ext_vector_type(16))) _Float16 v16h;
typedef __attribute__((ext_vector_type(8)))  float    v8f;

union FragH { v16h h; unsigned u[8]; };
union PackB { _Float16 h[8]; uint4 v; };

// ---------------------------------------------------------------------------
// Implicit-GEMM conv kernel.
//   Out[m][zoff + n] = sum_k A[m][k] * X(k, zoff+n) + bias[m]
// A: f16, zero-padded (Mp x Kp) -> staging is one unconditional b128/thread.
// X: fp32 CN activations; k-decode uses magic-multiply division (exact for
// gk < 2^12, divisor <= 49). OOB columns / padded k only need address-safe
// clamps (outputs masked / A rows zero); only conv zero-padding needs selects.
// Block: 512 threads = 16 waves; 64x64 C tile; K stepped 64 (2 WMMA / barrier).
// MODE: 0 = 1x1 GEMM, 1 = conv pad==0 (pre-padded input), 2 = conv pad>0.
// ---------------------------------------------------------------------------
__global__ __launch_bounds__(512)
void k_conv_gemm(const _Float16* __restrict__ A, int lda, long aZStride,
                 const float* __restrict__ X, long xRowStride,
                 const float* __restrict__ bias,
                 float* __restrict__ C, int ldc,
                 int M, int N, int K, int Kp, int zColStride,
                 int Hin, int Win, int OH, int OW,
                 int KH, int KW, int stride, int pad,
                 unsigned mKHKW, unsigned mKW) {
  __shared__ __attribute__((aligned(16))) _Float16 sA[64 * 72]; // [row][k]
  __shared__ __attribute__((aligned(16))) _Float16 sB[64 * 72]; // [n][k]

  const int tid  = threadIdx.x;
  const int wave = tid >> 5;
  const int lane = tid & 31;
  const int mi = wave >> 2;
  const int ni = wave & 3;
  const int tileM = blockIdx.y * 64;
  const int tileN = blockIdx.x * 64;
  const int zoff  = blockIdx.z * zColStride;
  const _Float16* Ap = A + (long)blockIdx.z * aZStride + (long)tileM * lda;

  // A staging: thread -> (row, 8-wide k chunk)
  const int ar = tid >> 3;
  const int ac = (tid & 7) * 8;
  // B staging: thread -> (n, 8-wide k chunk); adjacent lanes -> adjacent n
  const int nl = tid & 63;
  const int kq = (tid >> 6) * 8;

  const int KHKW = KH * KW;
  const int OHOW = OH * OW;
  const int HWin = Hin * Win;

  // ---- per-thread spatial decode for B, hoisted out of the K loop ----
  int gn = tileN + nl;
  if (gn >= N) gn = N - 1;                  // address-safe clamp only
  const int neff = gn + zoff;
  const int bb = neff / OHOW, pp = neff % OHOW;
  const int oh = pp / OW,     ow = pp % OW;
  const int ihs = oh * stride - pad;
  const int iws = ow * stride - pad;
  const long xbase = (long)bb * HWin;
  const int ih1 = min(max(ihs, 0), Hin - 1);
  const int iw1 = min(max(iws, 0), Win - 1);
  const long xoff1 = xbase + (long)ih1 * Win + iw1;   // 1x1 fast-path offset

  // fragment gather coords
  const int fm  = lane & 15;
  const int kbA = (lane >> 4) << 3;   // 0 or 8
  const int kbB = (lane >> 4) << 4;   // 0 or 16
  const _Float16* arow = &sA[(mi * 16 + fm) * 72];
  const _Float16* brow = &sB[(ni * 16 + fm) * 72];

  v8f acc = {};

  auto run = [&](auto modeC) {
    constexpr int MODE = decltype(modeC)::value;
    for (int k0 = 0; k0 < Kp; k0 += 64) {
      // ---- A: one b128 load + one b128 LDS store, no guards ----
      {
        uint4 av = *(const uint4*)(Ap + (long)ar * lda + (k0 + ac));
        *(uint4*)(&sA[ar * 72 + ac]) = av;
      }
      // ---- B: 8 gathers -> f16, one b128 LDS store ----
      {
        PackB hb;
        #pragma unroll
        for (int q = 0; q < 8; ++q) {
          int gk = k0 + kq + q;
          gk = min(gk, K - 1);              // address-safe (A is zero for k>=K)
          float v;
          if (MODE == 0) {
            v = X[(long)gk * xRowStride + xoff1];
          } else {
            unsigned ci = ((unsigned)gk * mKHKW) >> 22;
            int rs = gk - (int)ci * KHKW;
            unsigned rr = ((unsigned)rs * mKW) >> 22;
            int ss = rs - (int)rr * KW;
            int ih = ihs + (int)rr, iw = iws + (int)ss;
            if (MODE == 1) {
              v = X[(long)ci * xRowStride + xbase + (long)ih * Win + iw];
            } else {
              bool ok = (ih >= 0) && (ih < Hin) && (iw >= 0) && (iw < Win);
              int ihc = min(max(ih, 0), Hin - 1);
              int iwc = min(max(iw, 0), Win - 1);
              v = X[(long)ci * xRowStride + xbase + (long)ihc * Win + iwc];
              v = ok ? v : 0.0f;            // functional zero-padding
            }
          }
          hb.h[q] = (_Float16)v;
        }
        *(uint4*)(&sB[nl * 72 + kq]) = hb.v;
      }
      __syncthreads();

      // ---- two 16x16x32 WMMAs over the 64-deep K slab ----
      #pragma unroll
      for (int ks = 0; ks < 64; ks += 32) {
        FragH af, bf;
        #pragma unroll
        for (int v = 0; v < 8; ++v)
          af.u[v] = *(const unsigned*)(arow + ks + kbB + 2 * v);
        #pragma unroll
        for (int v = 0; v < 8; ++v)
          bf.u[v] = *(const unsigned*)(brow + ks + kbB + 2 * v);
        acc = __builtin_amdgcn_wmma_f32_16x16x32_f16(
            false, af.h, false, bf.h, (short)0, acc, false, false);
      }
      __syncthreads();
    }
  };

  if (KHKW == 1)      run(std::integral_constant<int, 0>{});
  else if (pad == 0)  run(std::integral_constant<int, 1>{});
  else                run(std::integral_constant<int, 2>{});

  // ---- write C (lane n = lane&15; VGPR v -> m = 8*(lane>>4)+v) ----
  const int n     = lane & 15;
  const int mbase = (lane >> 4) << 3;
  const int col   = tileN + ni * 16 + n;
  if (col < N) {
    const long ccol = (long)col + zoff;
    #pragma unroll
    for (int v = 0; v < 8; ++v) {
      int row = tileM + mi * 16 + mbase + v;
      if (row < M) {
        float val = acc[v];
        if (bias) val += bias[row];
        C[(long)row * ldc + ccol] = val;
      }
    }
  }
}

// ---------------------------------------------------------------------------
// Weight packing: fp32 (M x K) -> f16 zero-padded (Mp x Kp)
// ---------------------------------------------------------------------------
__global__ void k_pack_w(const float* __restrict__ W, _Float16* __restrict__ Y,
                         int M, int K, int Kp, long total) {
  long i = (long)blockIdx.x * blockDim.x + threadIdx.x;
  if (i >= total) return;
  int k = (int)(i % Kp); int m = (int)(i / Kp);
  float v = (m < M && k < K) ? W[(long)m * K + k] : 0.0f;
  Y[i] = (_Float16)v;
}

// deconv: w(Cin,Cout,3,3) -> f16 padded W2[o][ci*9+r*3+s] = w[ci][o][2-r][2-s]
__global__ void k_pack_deconv_w(const float* __restrict__ W, _Float16* __restrict__ Y,
                                int Cin, int Cout, int Kp, long total) {
  long i = (long)blockIdx.x * blockDim.x + threadIdx.x;
  if (i >= total) return;
  int k = (int)(i % Kp); int o = (int)(i / Kp);
  float v = 0.0f;
  if (k < Cin * 9 && o < Cout) {
    int ci = k / 9, rs = k % 9, r = rs / 3, s = rs % 3;
    v = W[((long)ci * Cout + o) * 9 + (2 - r) * 3 + (2 - s)];
  }
  Y[i] = (_Float16)v;
}

// ---------------------------------------------------------------------------
// Support kernels
// ---------------------------------------------------------------------------
__global__ void k_nchw_to_cn(const float* __restrict__ X, float* __restrict__ Y,
                             int B, int Cc, int HW) {
  long i = (long)blockIdx.x * blockDim.x + threadIdx.x;
  long total = (long)B * Cc * HW;
  if (i >= total) return;
  int p = (int)(i % HW); long t = i / HW;
  int b = (int)(t % B);  int c = (int)(t / B);
  Y[i] = X[((long)b * Cc + c) * HW + p];
}

__global__ void k_f32_to_f16(const float* __restrict__ X, _Float16* __restrict__ Y, long n) {
  long i = (long)blockIdx.x * blockDim.x + threadIdx.x;
  if (i < n) Y[i] = (_Float16)X[i];
}

__global__ void k_refpad(const float* __restrict__ X, float* __restrict__ Y,
                         int Cc, int B, int H, int W, int p) {
  int PH = H + 2 * p, PW = W + 2 * p;
  long total = (long)Cc * B * PH * PW;
  long i = (long)blockIdx.x * blockDim.x + threadIdx.x;
  if (i >= total) return;
  int pw = (int)(i % PW); long t = i / PW;
  int ph = (int)(t % PH); t /= PH;
  int b  = (int)(t % B);  int c = (int)(t / B);
  int h = ph - p; h = h < 0 ? -h : (h >= H ? 2 * H - 2 - h : h);
  int w = pw - p; w = w < 0 ? -w : (w >= W ? 2 * W - 2 - w : w);
  Y[i] = X[((long)c * B + b) * ((long)H * W) + (long)h * W + w];
}

// zero-stuffed upsample for ConvTranspose2d(k3,s2,p1,op1)
__global__ void k_upsample(const float* __restrict__ X, float* __restrict__ Y,
                           int Cc, int B, int H, int W) {
  int UH = 2 * H + 2, UW = 2 * W + 2;
  long total = (long)Cc * B * UH * UW;
  long i = (long)blockIdx.x * blockDim.x + threadIdx.x;
  if (i >= total) return;
  int uw = (int)(i % UW); long t = i / UW;
  int uh = (int)(t % UH); t /= UH;
  int b  = (int)(t % B);  int c = (int)(t / B);
  int dh = uh - 1, dw = uw - 1;
  float v = 0.0f;
  if (dh >= 0 && dh <= 2 * H - 2 && !(dh & 1) &&
      dw >= 0 && dw <= 2 * W - 2 && !(dw & 1))
    v = X[((long)c * B + b) * ((long)H * W) + (long)(dh >> 1) * W + (dw >> 1)];
  Y[i] = v;
}

__global__ void k_bn_stats(const float* __restrict__ X, long rowStride, int N,
                           float* __restrict__ mean, float* __restrict__ var) {
  int c = blockIdx.x;
  const float* row = X + (long)c * rowStride;
  float s = 0.0f, sq = 0.0f;
  for (int i = threadIdx.x; i < N; i += blockDim.x) {
    float v = row[i]; s += v; sq += v * v;
  }
  __shared__ float rs[256], rq[256];
  rs[threadIdx.x] = s; rq[threadIdx.x] = sq;
  __syncthreads();
  for (int off = 128; off > 0; off >>= 1) {
    if (threadIdx.x < off) { rs[threadIdx.x] += rs[threadIdx.x + off];
                             rq[threadIdx.x] += rq[threadIdx.x + off]; }
    __syncthreads();
  }
  if (threadIdx.x == 0) {
    float m = rs[0] / (float)N;
    mean[c] = m;
    var[c]  = rq[0] / (float)N - m * m;   // biased, matches reference
  }
}

__global__ void k_bn_act(const float* __restrict__ X, float* __restrict__ Y,
                         const float* __restrict__ mean, const float* __restrict__ var,
                         const float* __restrict__ sc, const float* __restrict__ bc,
                         long N, long total, int lrelu) {
  long i = (long)blockIdx.x * blockDim.x + threadIdx.x;
  if (i >= total) return;
  int c = (int)(i / N);
  float v = (X[i] - mean[c]) * rsqrtf(var[c] + 1e-5f) * sc[c] + bc[c];
  if (lrelu && v < 0.0f) v *= 0.1f;
  Y[i] = v;
}

// attn_logit[b,i,j], j=a*16+b2: sum_k F[k*256+i][b*256+b2*16+a] * G[k*256+i][b*256+a*16+b2]
__global__ void k_attn_logits(const float* __restrict__ F, const float* __restrict__ G,
                              float* __restrict__ L) {
  long i = (long)blockIdx.x * blockDim.x + threadIdx.x;
  if (i >= (long)16 * 256 * 256) return;
  int j  = (int)(i & 255); long t = i >> 8;
  int ii = (int)(t & 255); int b = (int)(t >> 8);
  int aq = j >> 4, b2 = j & 15;
  long baseF = (long)b * 256 + (b2 * 16 + aq);
  long baseG = (long)b * 256 + (aq * 16 + b2);
  float s = 0.0f;
  #pragma unroll 4
  for (int k = 0; k < 32; ++k) {
    long row = (long)(k * 256 + ii) * 4096;
    s += F[row + baseF] * G[row + baseG];
  }
  L[i] = s;
}

__global__ void k_softmax256(const float* __restrict__ X, float* __restrict__ Y) {
  int row = blockIdx.x;
  float v = X[(long)row * 256 + threadIdx.x];
  __shared__ float red[256];
  red[threadIdx.x] = v; __syncthreads();
  for (int off = 128; off > 0; off >>= 1) {
    if (threadIdx.x < off) red[threadIdx.x] = fmaxf(red[threadIdx.x], red[threadIdx.x + off]);
    __syncthreads();
  }
  float mx = red[0]; __syncthreads();
  float e = __expf(v - mx);
  red[threadIdx.x] = e; __syncthreads();
  for (int off = 128; off > 0; off >>= 1) {
    if (threadIdx.x < off) red[threadIdx.x] += red[threadIdx.x + off];
    __syncthreads();
  }
  Y[(long)row * 256 + threadIdx.x] = e / red[0];
}

// attn(B,256,256) -> CN transpose: T[j][b*256 + i] = attn[b][i][j]
__global__ void k_attn_T(const float* __restrict__ A, float* __restrict__ T) {
  long idx = (long)blockIdx.x * blockDim.x + threadIdx.x;
  if (idx >= (long)16 * 65536) return;
  int ii = (int)(idx % 256); long t = idx / 256;
  int b = (int)(t % 16); int j = (int)(t / 16);
  T[idx] = A[(long)b * 65536 + (long)ii * 256 + j];
}

__global__ void k_fuse_gamma(const float* __restrict__ AO, const float* __restrict__ Xr,
                             const float* __restrict__ gamma, float* __restrict__ Y, long n) {
  long i = (long)blockIdx.x * blockDim.x + threadIdx.x;
  if (i < n) Y[i] = gamma[0] * AO[i] + Xr[i];
}

__global__ void k_tanh_out(const float* __restrict__ X, float* __restrict__ Y, long n) {
  long i = (long)blockIdx.x * blockDim.x + threadIdx.x;
  if (i < n) Y[i] = tanhf(X[i]);
}

// ---------------------------------------------------------------------------
// Host orchestration
// ---------------------------------------------------------------------------
static inline size_t alignUp256(size_t x) { return (x + 255) & ~(size_t)255; }
static inline int ceil64(int x) { return (x + 63) & ~63; }
static inline dim3 ew(long n) { return dim3((unsigned)((n + 255) / 256)); }

extern "C" void kernel_launch(void* const* d_in, const int* in_sizes, int n_in,
                              void* d_out, int out_size, void* d_ws, size_t ws_size,
                              hipStream_t stream) {
  (void)in_sizes; (void)n_in; (void)out_size; (void)ws_size;

  const float* z     = (const float*)d_in[0];
  const float* c0_w  = (const float*)d_in[1];  const float* c0_b  = (const float*)d_in[2];
  const float* bn0_s = (const float*)d_in[3];  const float* bn0_b = (const float*)d_in[4];
  const float* c1_w  = (const float*)d_in[5];  const float* c1_b  = (const float*)d_in[6];
  const float* bn1_s = (const float*)d_in[7];  const float* bn1_b = (const float*)d_in[8];
  const float* c2_w  = (const float*)d_in[9];  const float* c2_b  = (const float*)d_in[10];
  const float* bn2_s = (const float*)d_in[11]; const float* bn2_b = (const float*)d_in[12];
  const float* af_w[2] = {(const float*)d_in[13], (const float*)d_in[20]};
  const float* af_b[2] = {(const float*)d_in[14], (const float*)d_in[21]};
  const float* ag_w[2] = {(const float*)d_in[15], (const float*)d_in[22]};
  const float* ag_b[2] = {(const float*)d_in[16], (const float*)d_in[23]};
  const float* ah_w[2] = {(const float*)d_in[17], (const float*)d_in[24]};
  const float* ah_b[2] = {(const float*)d_in[18], (const float*)d_in[25]};
  const float* agam[2] = {(const float*)d_in[19], (const float*)d_in[26]};
  const float* d0_w  = (const float*)d_in[43]; const float* d0_b  = (const float*)d_in[44];
  const float* bn7_s = (const float*)d_in[45]; const float* bn7_b = (const float*)d_in[46];
  const float* d1_w  = (const float*)d_in[47]; const float* d1_b  = (const float*)d_in[48];
  const float* bn8_s = (const float*)d_in[49]; const float* bn8_b = (const float*)d_in[50];
  const float* cl_w  = (const float*)d_in[51]; const float* cl_b  = (const float*)d_in[52];

  float* out_img = (float*)d_out;
  float* p1 = out_img + 65536;
  float* p2 = p1 + 1048576;

  char* ws = (char*)d_ws;
  size_t off = 0;
  auto alloc = [&](size_t bytes) -> void* {
    void* p = ws + off; off = alignUp256(off + bytes); return p;
  };
  float* zcn  = (float*)alloc((size_t)2 * 65536 * 4);
  float* pad0 = (float*)alloc((size_t)2 * 16 * 70 * 70 * 4);
  float* a0   = (float*)alloc((size_t)64 * 65536 * 4);        // also d1 output
  float* a1   = (float*)alloc((size_t)128 * 16384 * 4);       // also d0 output
  float* a2   = (float*)alloc((size_t)256 * 4096 * 4);
  float* r1   = (float*)alloc((size_t)256 * 4096 * 4);        // logits / res t1
  float* r2   = (float*)alloc((size_t)256 * 4096 * 4);        // hx
  float* r3   = (float*)alloc((size_t)256 * 4096 * 4);        // attnT (CN)
  float* r4   = (float*)alloc((size_t)256 * 4096 * 4);        // attn out
  float* r5   = (float*)alloc((size_t)256 * 4096 * 4);        // attn fused / res t2
  float* padr = (float*)alloc((size_t)256 * 16 * 18 * 18 * 4);
  float* Fm   = (float*)alloc((size_t)8192 * 4096 * 4);       // also ups0 / ups1
  float* Gm   = (float*)alloc((size_t)8192 * 4096 * 4);       // also padF
  _Float16* w16  = (_Float16*)alloc((size_t)8192 * 256 * 2);
  _Float16* hx16 = (_Float16*)alloc((size_t)256 * 4096 * 2);
  float* statM = (float*)alloc((size_t)8192 * 4);
  float* statV = (float*)alloc((size_t)8192 * 4);
  float* finb  = (float*)alloc((size_t)65536 * 4);

  auto gemm = [&](const _Float16* A, int lda, long aZ, const float* X, long xRow,
                  const float* bias, float* C, int ldc, int M, int N, int K, int Kp,
                  int Z, int zCol, int Hin, int Win, int OH, int OW,
                  int KH, int KW, int st, int pd) {
    dim3 g((N + 63) / 64, (M + 63) / 64, Z);
    unsigned d1 = (unsigned)(KH * KW), d2 = (unsigned)KW;
    unsigned mKHKW = (4194304u + d1 - 1) / d1;
    unsigned mKW   = (4194304u + d2 - 1) / d2;
    k_conv_gemm<<<g, 512, 0, stream>>>(A, lda, aZ, X, xRow, bias, C, ldc,
                                       M, N, K, Kp, zCol, Hin, Win, OH, OW,
                                       KH, KW, st, pd, mKHKW, mKW);
  };
  auto packW = [&](const float* w, int M, int K) -> int {
    int Kp = ceil64(K), Mp = ceil64(M);
    long tot = (long)Mp * Kp;
    k_pack_w<<<ew(tot), 256, 0, stream>>>(w, w16, M, K, Kp, tot);
    return Kp;
  };
  auto bn = [&](float* x, int C, int N, const float* s, const float* b, bool act) {
    k_bn_stats<<<C, 256, 0, stream>>>(x, (long)N, N, statM, statV);
    long tot = (long)C * N;
    k_bn_act<<<ew(tot), 256, 0, stream>>>(x, x, statM, statV, s, b, (long)N, tot, act ? 1 : 0);
  };

  // ---- stem ----
  k_nchw_to_cn<<<ew((long)2 * 65536), 256, 0, stream>>>(z, zcn, 16, 2, 4096);
  k_refpad<<<ew((long)2 * 16 * 70 * 70), 256, 0, stream>>>(zcn, pad0, 2, 16, 64, 64, 3);
  { int Kp = packW(c0_w, 64, 98);
    gemm(w16, Kp, 0, pad0, 78400, c0_b, a0, 65536, 64, 65536, 98, Kp, 1, 0,
         70, 70, 64, 64, 7, 7, 1, 0); }
  bn(a0, 64, 65536, bn0_s, bn0_b, true);

  { int Kp = packW(c1_w, 128, 576);
    gemm(w16, Kp, 0, a0, 65536, c1_b, a1, 16384, 128, 16384, 576, Kp, 1, 0,
         64, 64, 32, 32, 3, 3, 2, 1); }
  bn(a1, 128, 16384, bn1_s, bn1_b, true);

  { int Kp = packW(c2_w, 256, 1152);
    gemm(w16, Kp, 0, a1, 16384, c2_b, a2, 4096, 256, 4096, 1152, Kp, 1, 0,
         32, 32, 16, 16, 3, 3, 2, 1); }
  bn(a2, 256, 4096, bn2_s, bn2_b, true);

  // ---- self-attention ----
  auto attn = [&](const float* xin, float* xout, int li, float* attn_dst) {
    { int Kp = packW(af_w[li], 8192, 256);
      gemm(w16, Kp, 0, xin, 4096, af_b[li], Fm, 4096, 8192, 4096, 256, Kp, 1, 0,
           16, 16, 16, 16, 1, 1, 1, 0); }
    { int Kp = packW(ag_w[li], 8192, 256);
      gemm(w16, Kp, 0, xin, 4096, ag_b[li], Gm, 4096, 8192, 4096, 256, Kp, 1, 0,
           16, 16, 16, 16, 1, 1, 1, 0); }
    k_attn_logits<<<ew((long)16 * 65536), 256, 0, stream>>>(Fm, Gm, r1);
    k_softmax256<<<4096, 256, 0, stream>>>(r1, attn_dst);
    { int Kp = packW(ah_w[li], 256, 256);
      gemm(w16, Kp, 0, xin, 4096, ah_b[li], r2, 4096, 256, 4096, 256, Kp, 1, 0,
           16, 16, 16, 16, 1, 1, 1, 0); }
    k_f32_to_f16<<<ew((long)256 * 4096), 256, 0, stream>>>(r2, hx16, (long)256 * 4096);
    k_attn_T<<<ew((long)16 * 65536), 256, 0, stream>>>(attn_dst, r3);
    // batched: per-batch Out(256x256) = hx16[b](256x256) x attnT_cn[b](256x256)
    gemm(hx16, 4096, 256, r3, 4096, nullptr, r4, 4096, 256, 256, 256, 256, 16, 256,
         16, 16, 16, 16, 1, 1, 1, 0);
    k_fuse_gamma<<<ew((long)256 * 4096), 256, 0, stream>>>(r4, xin, agam[li], xout,
                                                           (long)256 * 4096);
  };

  attn(a2, r5, 0, p1);

  // ---- 2 conv blocks (no skip; matches reference) ----
  auto resconv = [&](const float* xin, float* xout, int pidx, bool act) {
    const float* wc  = (const float*)d_in[pidx + 0];
    const float* bc  = (const float*)d_in[pidx + 1];
    const float* bns = (const float*)d_in[pidx + 2];
    const float* bnb = (const float*)d_in[pidx + 3];
    k_refpad<<<ew((long)256 * 16 * 18 * 18), 256, 0, stream>>>(xin, padr, 256, 16, 16, 16, 1);
    int Kp = packW(wc, 256, 2304);
    gemm(w16, Kp, 0, padr, 5184, bc, xout, 4096, 256, 4096, 2304, Kp, 1, 0,
         18, 18, 16, 16, 3, 3, 1, 0);
    bn(xout, 256, 4096, bns, bnb, act);
  };
  resconv(r5, r1, 27, true);   // l2_0a
  resconv(r1, a2, 31, false);  // l2_0b
  resconv(a2, r1, 35, true);   // l2_1a
  resconv(r1, r5, 39, false);  // l2_1b

  attn(r5, a2, 1, p2);

  // ---- deconv 0: 256 -> 128, 16x16 -> 32x32 ----
  k_upsample<<<ew((long)256 * 16 * 34 * 34), 256, 0, stream>>>(a2, Fm, 256, 16, 16, 16);
  { int Kp = ceil64(256 * 9);
    long tot = (long)128 * Kp;
    k_pack_deconv_w<<<ew(tot), 256, 0, stream>>>(d0_w, w16, 256, 128, Kp, tot);
    gemm(w16, Kp, 0, Fm, 18496, d0_b, a1, 16384, 128, 16384, 2304, Kp, 1, 0,
         34, 34, 32, 32, 3, 3, 1, 0); }
  bn(a1, 128, 16384, bn7_s, bn7_b, true);

  // ---- deconv 1: 128 -> 64, 32x32 -> 64x64 ----
  k_upsample<<<ew((long)128 * 16 * 66 * 66), 256, 0, stream>>>(a1, Fm, 128, 16, 32, 32);
  { int Kp = ceil64(128 * 9);
    long tot = (long)64 * Kp;
    k_pack_deconv_w<<<ew(tot), 256, 0, stream>>>(d1_w, w16, 128, 64, Kp, tot);
    gemm(w16, Kp, 0, Fm, 69696, d1_b, a0, 65536, 64, 65536, 1152, Kp, 1, 0,
         66, 66, 64, 64, 3, 3, 1, 0); }
  bn(a0, 64, 65536, bn8_s, bn8_b, true);

  // ---- final: refpad 3, 7x7 conv to 1 channel, tanh ----
  k_refpad<<<ew((long)64 * 16 * 70 * 70), 256, 0, stream>>>(a0, Gm, 64, 16, 64, 64, 3);
  { int Kp = packW(cl_w, 1, 3136);
    gemm(w16, Kp, 0, Gm, 78400, cl_b, finb, 65536, 1, 65536, 3136, Kp, 1, 0,
         70, 70, 64, 64, 7, 7, 1, 0); }
  k_tanh_out<<<ew(65536), 256, 0, stream>>>(finb, out_img, 65536);
}
